// FCPairedLayer_67087389164340
// MI455X (gfx1250) — matrix operand order repaired and checked by reading, not compiled, not run on hardware
//
#include <hip/hip_runtime.h>

typedef __attribute__((ext_vector_type(16))) _Float16 v16h;
typedef __attribute__((ext_vector_type(8)))  _Float16 v8h;
typedef __attribute__((ext_vector_type(8)))  float    v8f;
typedef _Float16 f16;

#define NROW 1024   // B*N
#define CIN  64
#define NH   256
#define NHQ  64     // NH/4 float4 chunks
#define NF   384
#define NN   512

// Workspace layout (bytes):
//   Xh  : NROW*CIN f16              @ 0         (131072 B)
//   W1T : NH*NF f16 (transposed)    @ 131072    (196608 B)
//   G   : 6*NROW*NH f32             @ 327680    (6291456 B)
//   A   : NROW*NH f32               @ 6619136   (1048576 B)
//   Bv  : NROW*NH f32               @ 7667712   (1048576 B)
// total ~8.7 MB

__device__ __forceinline__ float4 f4add(float4 a, float4 b) {
    return make_float4(a.x + b.x, a.y + b.y, a.z + b.z, a.w + b.w);
}
__device__ __forceinline__ float reludot(float4 v, float4 w) {
    return fmaxf(v.x, 0.f) * w.x + fmaxf(v.y, 0.f) * w.y +
           fmaxf(v.z, 0.f) * w.z + fmaxf(v.w, 0.f) * w.w;
}

// ---------------------------------------------------------------------------
// Kernel 1: f32 -> f16 conversions (X as-is, W1 transposed to (h, c) layout)
// ---------------------------------------------------------------------------
__global__ void fcpair_prep_kernel(const float* __restrict__ x,
                                   const float* __restrict__ W1,
                                   f16* __restrict__ Xh,
                                   f16* __restrict__ W1T) {
    const int total = NROW * CIN + NH * NF;
    for (int t = blockIdx.x * blockDim.x + threadIdx.x; t < total;
         t += gridDim.x * blockDim.x) {
        if (t < NROW * CIN) {
            Xh[t] = (f16)x[t];
        } else {
            int u = t - NROW * CIN;          // u = h*NF + c
            int h = u / NF, c = u - h * NF;
            W1T[u] = (f16)W1[c * NH + h];    // W1 is (384, 256) row-major
        }
    }
}

// ---------------------------------------------------------------------------
// Kernel 2: G[k][row][h] = X[row, :] @ W1[k*64:(k+1)*64, :]  via WMMA f16->f32
// One wave per 16x16 output tile; K=64 -> two v_wmma_f32_16x16x32_f16.
// ---------------------------------------------------------------------------
__global__ void fcpair_gemm_wmma_kernel(const f16* __restrict__ Xh,
                                        const f16* __restrict__ W1T,
                                        float* __restrict__ G) {
    const int wid  = blockIdx.x * (blockDim.x >> 5) + (threadIdx.x >> 5);
    const int lane = threadIdx.x & 31;
    // tiles: mT in [0,64) x k in [0,6) x hT in [0,16)  -> 6144 waves
    const int mT  = wid / 96;
    const int rem = wid - mT * 96;
    const int k   = rem >> 4;
    const int hT  = rem & 15;
    if (mT >= 64) return;

    const int half = lane >> 4;   // 0: lanes 0-15, 1: lanes 16-31
    const int l16  = lane & 15;

    const f16* arow = Xh  + (mT * 16 + l16) * CIN;          // A row M = l16
    const f16* brow = W1T + (hT * 16 + l16) * NF + k * CIN; // B col N = l16

    v8f c = {};
#pragma unroll
    for (int s = 0; s < 2; ++s) {   // K-steps of 32
        // A 16x32 f16 layout: half0 -> K{0..7,16..23}, half1 -> K{8..15,24..31}
        v8h a0 = *(const v8h*)(arow + s * 32 + half * 8);
        v8h a1 = *(const v8h*)(arow + s * 32 + 16 + half * 8);
        v16h av = __builtin_shufflevector(a0, a1,
                    0,1,2,3,4,5,6,7,8,9,10,11,12,13,14,15);
        // B 32x16 f16 layout: half0 -> K{0..7,16..23}, half1 -> K{8..15,24..31}
        v8h b0 = *(const v8h*)(brow + s * 32 + half * 8);
        v8h b1v = *(const v8h*)(brow + s * 32 + 16 + half * 8);
        v16h bv = __builtin_shufflevector(b0, b1v,
                    0,1,2,3,4,5,6,7,8,9,10,11,12,13,14,15);
        c = __builtin_amdgcn_wmma_f32_16x16x32_f16(
                /*neg_a=*/false, av, /*neg_b=*/false, bv,
                /*c_mod=*/(short)0, c, /*reuse_a=*/false, /*reuse_b=*/false);
    }
    // C/D layout: lane = n + 16*(m/8), VGPR r -> m = 8*half + r
    float* gout = G + (size_t)(k * NROW + mT * 16 + half * 8) * NH + hT * 16 + l16;
#pragma unroll
    for (int r = 0; r < 8; ++r) gout[r * NH] = c[r];
}

// ---------------------------------------------------------------------------
// Kernel 3: per-row combined tables (valid for interior pairs):
//   A[row]  = G0[row] + G2[row-1] + G4[row+1] + b1
//   Bv[row] = G1[row] + G3[row+1] + G5[row-1]
// Edge rows get guarded (border pairs are fixed up exactly later anyway).
// ---------------------------------------------------------------------------
__global__ void fcpair_rowcomb_kernel(const float* __restrict__ G,
                                      const float* __restrict__ b1,
                                      float* __restrict__ A,
                                      float* __restrict__ Bv) {
    const int t = blockIdx.x * blockDim.x + threadIdx.x;  // 65536 float4 slots
    if (t >= NROW * NHQ) return;
    const int row = t >> 6;         // /NHQ
    const int hq  = t & (NHQ - 1);
    const int i   = row & (NN - 1); // index within batch

    const float4* G0 = (const float4*)G;
    const float4* G1 = G0 + 1 * NROW * NHQ;
    const float4* G2 = G0 + 2 * NROW * NHQ;
    const float4* G3 = G0 + 3 * NROW * NHQ;
    const float4* G4 = G0 + 4 * NROW * NHQ;
    const float4* G5 = G0 + 5 * NROW * NHQ;
    const float4* b1q = (const float4*)b1;

    float4 a = f4add(G0[row * NHQ + hq], b1q[hq]);
    if (i >= 1)      a = f4add(a, G2[(row - 1) * NHQ + hq]);
    if (i <= NN - 2) a = f4add(a, G4[(row + 1) * NHQ + hq]);

    float4 bb = G1[row * NHQ + hq];
    if (i <= NN - 2) bb = f4add(bb, G3[(row + 1) * NHQ + hq]);
    if (i >= 1)      bb = f4add(bb, G5[(row - 1) * NHQ + hq]);

    ((float4*)A)[t]  = a;
    ((float4*)Bv)[t] = bb;
}

// ---------------------------------------------------------------------------
// Kernel 4: fast combine for all pairs: y = relu(A[i]+Bv[j])·W2 + b2.
// Block = 16 i x 32 j tile fully staged in LDS; each thread -> 2 outputs.
// B tile padded to 65-float4 row stride for conflict-free LDS reads.
// ---------------------------------------------------------------------------
#define BPAD 65
__global__ void fcpair_combine_kernel(const float* __restrict__ A,
                                      const float* __restrict__ Bv,
                                      const float* __restrict__ W2,
                                      const float* __restrict__ b2,
                                      float* __restrict__ out) {
    __shared__ float4 sA[16 * BPAD];
    __shared__ float4 sB[32 * BPAD];
    __shared__ float4 sW[NHQ];

    const int tid = threadIdx.x;            // 256 threads
    const int blk = blockIdx.x;             // 2 * 32 * 16
    const int jT = blk & 15;
    const int iT = (blk >> 4) & 31;
    const int b  = blk >> 9;

    const float4* Aq = (const float4*)A;
    const float4* Bq = (const float4*)Bv;

    for (int t = tid; t < 16 * NHQ; t += 256) {
        int r = t >> 6, hq = t & (NHQ - 1);
        sA[r * BPAD + hq] = Aq[(size_t)(b * NN + iT * 16 + r) * NHQ + hq];
    }
    for (int t = tid; t < 32 * NHQ; t += 256) {
        int r = t >> 6, hq = t & (NHQ - 1);
        sB[r * BPAD + hq] = Bq[(size_t)(b * NN + jT * 32 + r) * NHQ + hq];
    }
    if (tid < NHQ) sW[tid] = ((const float4*)W2)[tid];
    __syncthreads();

    const int ti = tid >> 4, tj = tid & 15;
    const int i  = iT * 16 + ti;
    const int j0 = jT * 32 + tj;
    const int j1 = j0 + 16;

    float acc0 = 0.f, acc1 = 0.f;
#pragma unroll 4
    for (int hq = 0; hq < NHQ; ++hq) {
        float4 a  = sA[ti * BPAD + hq];
        float4 w  = sW[hq];
        float4 q0 = sB[tj * BPAD + hq];
        float4 q1 = sB[(tj + 16) * BPAD + hq];
        acc0 += reludot(f4add(a, q0), w);
        acc1 += reludot(f4add(a, q1), w);
    }
    const float bias = b2[0];
    float* orow = out + (size_t)(b * NN + i) * NN;
    orow[j0] = acc0 + bias;
    orow[j1] = acc1 + bias;
}

// ---------------------------------------------------------------------------
// Kernel 5: exact masked recompute for border pairs (i or j in {0, 511}).
// Overwrites the fast-path values; ~4K pairs total.
// ---------------------------------------------------------------------------
__global__ void fcpair_edge_kernel(const float* __restrict__ G,
                                   const float* __restrict__ b1,
                                   const float* __restrict__ W2,
                                   const float* __restrict__ b2,
                                   float* __restrict__ out) {
    const int t = blockIdx.x * blockDim.x + threadIdx.x;  // 2 * 4 * 512
    if (t >= 2 * 4 * NN) return;
    const int b    = t >> 11;
    const int r    = t & 2047;
    const int side = r >> 9;
    const int idx  = r & (NN - 1);

    int i, j;
    if      (side == 0) { i = 0;      j = idx; }
    else if (side == 1) { i = NN - 1; j = idx; }
    else if (side == 2) { i = idx;    j = 0;  if (idx == 0 || idx == NN - 1) return; }
    else                { i = idx;    j = NN - 1; if (idx == 0 || idx == NN - 1) return; }

    const int rowi = b * NN + i;
    const int rowj = b * NN + j;
    const bool mu = (i >= 1) && (j <= NN - 2);
    const bool md = (i <= NN - 2) && (j >= 1);

    const float4* G0 = (const float4*)G;
    const float4* G1 = G0 + 1 * NROW * NHQ;
    const float4* G2 = G0 + 2 * NROW * NHQ;
    const float4* G3 = G0 + 3 * NROW * NHQ;
    const float4* G4 = G0 + 4 * NROW * NHQ;
    const float4* G5 = G0 + 5 * NROW * NHQ;
    const float4* b1q = (const float4*)b1;
    const float4* W2q = (const float4*)W2;

    float acc = 0.f;
    for (int hq = 0; hq < NHQ; ++hq) {
        float4 v = f4add(G0[rowi * NHQ + hq], G1[rowj * NHQ + hq]);
        v = f4add(v, b1q[hq]);
        if (mu) {
            v = f4add(v, G2[(rowi - 1) * NHQ + hq]);
            v = f4add(v, G3[(rowj + 1) * NHQ + hq]);
        }
        if (md) {
            v = f4add(v, G4[(rowi + 1) * NHQ + hq]);
            v = f4add(v, G5[(rowj - 1) * NHQ + hq]);
        }
        acc += reludot(v, W2q[hq]);
    }
    out[(size_t)rowi * NN + j] = acc + b2[0];
}

// ---------------------------------------------------------------------------
extern "C" void kernel_launch(void* const* d_in, const int* in_sizes, int n_in,
                              void* d_out, int out_size, void* d_ws, size_t ws_size,
                              hipStream_t stream) {
    const float* x  = (const float*)d_in[0];   // (2,512,64)
    const float* W1 = (const float*)d_in[1];   // (384,256)
    const float* b1 = (const float*)d_in[2];   // (256,)
    const float* W2 = (const float*)d_in[3];   // (256,1)
    const float* b2 = (const float*)d_in[4];   // (1,)
    float* out = (float*)d_out;                // (2,512,512,1)

    char*  ws  = (char*)d_ws;
    f16*   Xh  = (f16*)ws;                     // 131072 B
    f16*   W1T = (f16*)(ws + 131072);          // 196608 B
    float* G   = (float*)(ws + 327680);        // 6291456 B
    float* A   = (float*)(ws + 6619136);       // 1048576 B
    float* Bv  = (float*)(ws + 7667712);       // 1048576 B

    fcpair_prep_kernel<<<640, 256, 0, stream>>>(x, W1, Xh, W1T);
    fcpair_gemm_wmma_kernel<<<768, 256, 0, stream>>>(Xh, W1T, G);   // 6144 waves
    fcpair_rowcomb_kernel<<<256, 256, 0, stream>>>(G, b1, A, Bv);
    fcpair_combine_kernel<<<1024, 256, 0, stream>>>(A, Bv, W2, b2, out);
    fcpair_edge_kernel<<<16, 256, 0, stream>>>(G, b1, W2, b2, out);
}
